// Model_8650064134637
// MI455X (gfx1250) — matrix operand order repaired and checked by reading, not compiled, not run on hardware
//
#include <hip/hip_runtime.h>

// Transposed RNN scan on gfx1250, permuted-hidden-state formulation.
//   state B-fragment K-slot k holds hidden unit tau(k), tau swaps groups 1<->2 (of 8).
//   A = W_hh with tau-permuted columns => D-tile locals ARE the next B fragment:
//   no cross-lane exchange, no selects. f16-accumulator WMMA => no cvt in the loop.
//   ReLU = one v_pk_max_num_f16 per dword (inline asm so LLVM can't expand it).

typedef __attribute__((ext_vector_type(16))) _Float16 v16h;
typedef __attribute__((ext_vector_type(8)))  _Float16 v8h;
typedef __attribute__((ext_vector_type(2)))  _Float16 v2h;
typedef __attribute__((ext_vector_type(8)))  float    v8f;

#define SEQ_L 512
#define HID   32
#define NCLS  4

#if __has_builtin(__builtin_amdgcn_wmma_f16_16x16x32_f16)
#define USE_F16_WMMA 1
#endif

union B8  { v8h  v; v2h p[4]; int i[4]; };
union B16 { v16h v; v2h p[8]; int i[8]; };

__device__ __forceinline__ int pk_relu(int a) {
    int r;
    asm("v_pk_max_num_f16 %0, %1, 0" : "=v"(r) : "v"(a));
    return r;
}

struct Consts {
    v16h Ah0, Ah1;                 // W_hh A-fragments (tau-permuted columns)
    B8   w2A, w2B, bb2A, bb2B;     // packed f16 input-projection constants
};

__device__ __forceinline__ v16h rnn_step(float xv, v16h st, const Consts& cc)
{
    const _Float16 xh = (_Float16)xv;
    v2h x2; x2[0] = xh; x2[1] = xh;

    // C tiles, f16 C-layout: dword v = hidden pair (2v + 8*hg [+16 for tile1])
    B8 c0, c1;
#pragma unroll
    for (int v = 0; v < 4; ++v) {
        c0.p[v] = x2 * cc.w2A.p[v] + cc.bb2A.p[v];   // v_pk_fma_f16
        c1.p[v] = x2 * cc.w2B.p[v] + cc.bb2B.p[v];
    }

    B8 d0, d1;
#ifdef USE_F16_WMMA
    d0.v = __builtin_amdgcn_wmma_f16_16x16x32_f16(
               false, cc.Ah0, false, st, (short)0, c0.v, false, false);
    d1.v = __builtin_amdgcn_wmma_f16_16x16x32_f16(
               false, cc.Ah1, false, st, (short)0, c1.v, false, false);
#else
    v8f cf0, cf1;
#pragma unroll
    for (int v = 0; v < 8; ++v) {
        cf0[v] = (float)c0.p[v >> 1][v & 1];
        cf1[v] = (float)c1.p[v >> 1][v & 1];
    }
    v8f e0 = __builtin_amdgcn_wmma_f32_16x16x32_f16(
                 false, cc.Ah0, false, st, (short)0, cf0, false, false);
    v8f e1 = __builtin_amdgcn_wmma_f32_16x16x32_f16(
                 false, cc.Ah1, false, st, (short)0, cf1, false, false);
#pragma unroll
    for (int v = 0; v < 4; ++v) {
        v2h q0, q1;
        q0[0] = (_Float16)e0[2 * v];  q0[1] = (_Float16)e0[2 * v + 1];
        q1[0] = (_Float16)e1[2 * v];  q1[1] = (_Float16)e1[2 * v + 1];
        d0.p[v] = q0;  d1.p[v] = q1;
    }
#endif

    // D locals -> next B fragment directly (tau makes this the identity relayout)
    B16 nb;
#pragma unroll
    for (int v = 0; v < 4; ++v) {
        nb.i[v]     = pk_relu(d0.i[v]);   // K group: 0-7  | 16-23
        nb.i[v + 4] = pk_relu(d1.i[v]);   // K group: 8-15 | 24-31
    }
    return nb.v;
}

__global__ __launch_bounds__(32)
void rnn_scan_wmma(const float* __restrict__ x,      // [N, 512]
                   const float* __restrict__ W_ih,   // [32, 1]
                   const float* __restrict__ b_ih,   // [32]
                   const float* __restrict__ W_hh,   // [32, 32]
                   const float* __restrict__ b_hh,   // [32]
                   const float* __restrict__ W_out,  // [4, 32]
                   const float* __restrict__ b_out,  // [4]
                   float* __restrict__ out)          // [N, 4]
{
    __shared__ _Float16 Hst[2 * 16 * HID];   // final-state scratch for head, 2 KB

    const int lane = threadIdx.x;            // one wave per block
    const int m    = lane & 15;
    const int hg   = lane >> 4;              // half-lane group
    const int r0   = blockIdx.x * 32;        // 2 batch tiles of 16 rows

    Consts cc;
    // A-fragments: tau-permuted columns: col = e + 8*hg + 8*(e>>3)
#pragma unroll
    for (int e = 0; e < 16; ++e) {
        cc.Ah0[e] = (_Float16)W_hh[m * HID + e + 8 * hg + 8 * (e >> 3)];
        cc.Ah1[e] = (_Float16)W_hh[(m + 16) * HID + e + 8 * hg + 8 * (e >> 3)];
    }
    // packed input-projection constants: dword v = hidden pair (2v+8hg [+16])
#pragma unroll
    for (int v = 0; v < 4; ++v) {
        const int h0 = 2 * v + 8 * hg;
        const int h1 = h0 + 16;
        v2h w0, w1, b0, b1;
        w0[0] = (_Float16)W_ih[h0];     w0[1] = (_Float16)W_ih[h0 + 1];
        w1[0] = (_Float16)W_ih[h1];     w1[1] = (_Float16)W_ih[h1 + 1];
        b0[0] = (_Float16)(b_ih[h0] + b_hh[h0]);
        b0[1] = (_Float16)(b_ih[h0 + 1] + b_hh[h0 + 1]);
        b1[0] = (_Float16)(b_ih[h1] + b_hh[h1]);
        b1[1] = (_Float16)(b_ih[h1 + 1] + b_hh[h1 + 1]);
        cc.w2A.p[v] = w0;  cc.bb2A.p[v] = b0;
        cc.w2B.p[v] = w1;  cc.bb2B.p[v] = b1;
    }

    const float* px0 = x + (size_t)(r0 + m) * SEQ_L;        // tile 0 batch row
    const float* px1 = x + (size_t)(r0 + 16 + m) * SEQ_L;   // tile 1 batch row

    v16h st0 = {};   // h_{-1} = 0 => step 0 yields relu(x0*w_ih + b_ih + b_hh)
    v16h st1 = {};
    float4 xc0 = *(const float4*)px0;
    float4 xc1 = *(const float4*)px1;

    for (int tb = 0; tb < SEQ_L; tb += 4) {
        const int tp = (tb + 4 <= SEQ_L - 4) ? (tb + 4) : (SEQ_L - 4);
        float4 xn0 = *(const float4*)(px0 + tp);   // prefetch next 4 steps
        float4 xn1 = *(const float4*)(px1 + tp);
        const float xs0[4] = {xc0.x, xc0.y, xc0.z, xc0.w};
        const float xs1[4] = {xc1.x, xc1.y, xc1.z, xc1.w};
#pragma unroll
        for (int s = 0; s < 4; ++s) {
            st0 = rnn_step(xs0[s], st0, cc);   // two independent chains -> ILP
            st1 = rnn_step(xs1[s], st1, cc);
        }
        xc0 = xn0;  xc1 = xn1;
    }

    // ---- head: out = h_final @ W_out^T + b_out ----
    // state dword v holds hidden pair h = 2v + 8*((v>>2) + hg)  (un-permute tau)
    B16 u0, u1;  u0.v = st0;  u1.v = st1;
#pragma unroll
    for (int v = 0; v < 8; ++v) {
        const int h = 2 * v + 8 * ((v >> 2) + hg);
        *(v2h*)&Hst[0 * 16 * HID + m * HID + h] = u0.p[v];
        *(v2h*)&Hst[1 * 16 * HID + m * HID + h] = u1.p[v];
    }
    __syncthreads();

    const int cb = hg * 2;               // lanes 0-15: classes 0,1; 16-31: 2,3
    float a00 = b_out[cb], a01 = b_out[cb + 1];
    float a10 = b_out[cb], a11 = b_out[cb + 1];
#pragma unroll
    for (int h = 0; h < HID; ++h) {
        const float w0 = W_out[cb * HID + h];
        const float w1 = W_out[(cb + 1) * HID + h];
        const float h0 = (float)Hst[m * HID + h];
        const float h1 = (float)Hst[16 * HID + m * HID + h];
        a00 += h0 * w0;  a01 += h0 * w1;
        a10 += h1 * w0;  a11 += h1 * w1;
    }
    out[(size_t)(r0 + m) * NCLS + cb]          = a00;
    out[(size_t)(r0 + m) * NCLS + cb + 1]      = a01;
    out[(size_t)(r0 + 16 + m) * NCLS + cb]     = a10;
    out[(size_t)(r0 + 16 + m) * NCLS + cb + 1] = a11;
}

extern "C" void kernel_launch(void* const* d_in, const int* in_sizes, int n_in,
                              void* d_out, int out_size, void* d_ws, size_t ws_size,
                              hipStream_t stream) {
    const float* x     = (const float*)d_in[0];
    const float* W_ih  = (const float*)d_in[1];
    const float* b_ih  = (const float*)d_in[2];
    const float* W_hh  = (const float*)d_in[3];
    const float* b_hh  = (const float*)d_in[4];
    const float* W_out = (const float*)d_in[5];
    const float* b_out = (const float*)d_in[6];

    const int N = in_sizes[0] / SEQ_L;    // 8192
    dim3 grid(N / 32);                    // one wave, two 16-row tiles per block
    dim3 block(32);
    rnn_scan_wmma<<<grid, block, 0, stream>>>(x, W_ih, b_ih, W_hh, b_hh,
                                              W_out, b_out, (float*)d_out);
}
